// DiffusionLayer_19353122636426
// MI455X (gfx1250) — matrix operand order repaired and checked by reading, not compiled, not run on hardware
//
#include <hip/hip_runtime.h>
#include <math.h>

// ---------------------------------------------------------------------------
// CDNA5 (gfx1250) GATv2 diffusion layer.
// Dense projections: V_WMMA_F32_16X16X32_BF16 with pre-converted bf16 A and
// pre-packed fragment-order bf16 B (no conversions in the WMMA loop).
// Edge softmax/aggregation: atomics (segment max / sum / scatter-add).
// ---------------------------------------------------------------------------

typedef __attribute__((ext_vector_type(16))) __bf16    v16bf;
typedef __attribute__((ext_vector_type(8)))  float     v8f;
typedef __attribute__((ext_vector_type(4)))  unsigned  v4u;

union BFVec {
    v16bf v;
    v4u   q[2];          // two 16-byte chunks (8 bf16 each)
};

__device__ __forceinline__ unsigned short f2bf(float x) {
    unsigned b = __float_as_uint(x);
    b += 0x7FFFu + ((b >> 16) & 1u);     // round-to-nearest-even
    return (unsigned short)(b >> 16);
}

// Monotonic float <-> uint mapping for atomic float max
__device__ __forceinline__ unsigned f2ord(float f) {
    unsigned b = __float_as_uint(f);
    return (b & 0x80000000u) ? ~b : (b | 0x80000000u);
}
__device__ __forceinline__ float ord2f(unsigned u) {
    unsigned b = (u & 0x80000000u) ? (u & 0x7FFFFFFFu) : ~u;
    return __uint_as_float(b);
}

// ---------------------------------------------------------------------------
// f32 [N] -> bf16-bits [N] (row-major A operand for WMMA)
// ---------------------------------------------------------------------------
__global__ void cvt_f32_bf16(const float* __restrict__ in, unsigned short* __restrict__ out,
                             long long n) {
    long long i = (long long)blockIdx.x * blockDim.x + threadIdx.x;
    if (i < n) out[i] = f2bf(in[i]);
}

// ---------------------------------------------------------------------------
// Pack a 128x128 f32 row-major W into WMMA-B fragment order (bf16 bits):
// out[(((ks*8)+tileN)*32 + lane)*16 + i] = bf16(W[ks*32 + (lane>>4)*16 + i][tileN*16 + (lane&15)])
// 1024 threads total.
// ---------------------------------------------------------------------------
__global__ void pack_w_bf16(const float* __restrict__ W, unsigned short* __restrict__ out) {
    const int t = blockIdx.x * blockDim.x + threadIdx.x;
    if (t >= 1024) return;
    const int lane = t & 31;
    const int tn   = (t >> 5) & 7;
    const int ks   = t >> 8;
    const int half = lane >> 4;
    const int col  = tn * 16 + (lane & 15);
    const int kk   = ks * 32;
    unsigned short* o = out + (size_t)t * 16;
#pragma unroll
    for (int i = 0; i < 16; ++i)
        o[i] = f2bf(W[(size_t)(kk + half * 16 + i) * 128 + col]);
}

// ---------------------------------------------------------------------------
// C[M x 128] = A0 @ W0 (+ A1 @ W1) + bias. A: row-major bf16 bits [M x 128].
// B: fragment-packed bf16 bits (pack_w_bf16). One wave per 16x16 tile,
// grid = (M/16, 8). Inner loop: 4x b128 loads + 1 v_wmma per k-step.
// ---------------------------------------------------------------------------
__global__ __launch_bounds__(32)
void gemm_wmma_bf16(const unsigned short* __restrict__ A0, const unsigned short* __restrict__ B0,
                    const unsigned short* __restrict__ A1, const unsigned short* __restrict__ B1,
                    const float* __restrict__ bias, float* __restrict__ C) {
    const int tileM = blockIdx.x, tileN = blockIdx.y;
    const int lane  = threadIdx.x & 31;
    const int l15   = lane & 15;
    const int half  = lane >> 4;
    const int row   = tileM * 16 + l15;
    const int col   = tileN * 16 + l15;

    v8f acc = {};
    for (int pass = 0; pass < 2; ++pass) {
        const unsigned short* A = pass ? A1 : A0;
        const unsigned short* B = pass ? B1 : B0;
        if (A == nullptr) break;                     // uniform across the wave
        const unsigned short* arow = A + (size_t)row * 128;
#pragma unroll
        for (int ks = 0; ks < 4; ++ks) {
            const int kk = ks * 32;
            BFVec a, b;
            a.q[0] = *(const v4u*)(arow + kk + half * 16);
            a.q[1] = *(const v4u*)(arow + kk + half * 16 + 8);
            const unsigned short* bp = B + ((size_t)(ks * 8 + tileN) * 32 + lane) * 16;
            b.q[0] = *(const v4u*)(bp);
            b.q[1] = *(const v4u*)(bp + 8);
            acc = __builtin_amdgcn_wmma_f32_16x16x32_bf16(
                      false, a.v, false, b.v, (short)0, acc, false, false);
        }
    }
    const float bv = bias ? bias[col] : 0.0f;
#pragma unroll
    for (int v = 0; v < 8; ++v) {
        const int r = tileM * 16 + v + half * 8;
        C[(size_t)r * 128 + col] = acc[v] + bv;
    }
}

// ---------------------------------------------------------------------------
// e[edge] = attn . leaky_relu(fsrc[src] + fdst[dst], 0.2); segment max via
// ordered-uint atomicMax. One wave (32 lanes x 4 floats) per edge.
// ---------------------------------------------------------------------------
__global__ void edge_score(const float* __restrict__ fsrc, const float* __restrict__ fdst,
                           const int* __restrict__ src, const int* __restrict__ dst,
                           const float* __restrict__ attn,
                           float* __restrict__ e, unsigned* __restrict__ emax_ord, int E) {
    const int edge = blockIdx.x * 8 + (threadIdx.x >> 5);
    const int lane = threadIdx.x & 31;
    if (edge >= E) return;
    const int s = src[edge], d = dst[edge];
    const float4 fs = *(const float4*)(fsrc + (size_t)s * 128 + lane * 4);
    const float4 fd = *(const float4*)(fdst + (size_t)d * 128 + lane * 4);
    const float4 at = *(const float4*)(attn + lane * 4);
    float p = 0.f, x;
    x = fs.x + fd.x; x = x > 0.f ? x : 0.2f * x; p += x * at.x;
    x = fs.y + fd.y; x = x > 0.f ? x : 0.2f * x; p += x * at.y;
    x = fs.z + fd.z; x = x > 0.f ? x : 0.2f * x; p += x * at.z;
    x = fs.w + fd.w; x = x > 0.f ? x : 0.2f * x; p += x * at.w;
    for (int o = 16; o > 0; o >>= 1) p += __shfl_xor(p, o, 32);
    if (lane == 0) {
        e[edge] = p;
        atomicMax(emax_ord + d, f2ord(p));
    }
}

// ex = exp(e - emax[dst]); denom[dst] += ex.  One thread per edge.
__global__ void edge_exp(const float* __restrict__ e, const int* __restrict__ dst,
                         const unsigned* __restrict__ emax_ord,
                         float* __restrict__ ex, float* __restrict__ denom, int E) {
    const int i = blockIdx.x * blockDim.x + threadIdx.x;
    if (i >= E) return;
    const int d = dst[i];
    const unsigned u = emax_ord[d];
    const float m = (u == 0u) ? 0.0f : ord2f(u);   // empty segment -> 0 per reference
    const float v = __expf(e[i] - m);
    ex[i] = v;
    atomicAdd(denom + d, v);
}

// out[dst] += (ex/denom[dst]) * fsrc[src].  One wave per edge.
__global__ void edge_agg(const float* __restrict__ ex, const float* __restrict__ denom,
                         const float* __restrict__ fsrc,
                         const int* __restrict__ src, const int* __restrict__ dst,
                         float* __restrict__ out, int E) {
    const int edge = blockIdx.x * 8 + (threadIdx.x >> 5);
    const int lane = threadIdx.x & 31;
    if (edge >= E) return;
    const int s = src[edge], d = dst[edge];
    const float alpha = ex[edge] / denom[d];
    const float4 fs = *(const float4*)(fsrc + (size_t)s * 128 + lane * 4);
    float* o = out + (size_t)d * 128 + lane * 4;
    atomicAdd(o + 0, alpha * fs.x);
    atomicAdd(o + 1, alpha * fs.y);
    atomicAdd(o + 2, alpha * fs.z);
    atomicAdd(o + 3, alpha * fs.w);
}

// s[row] = leaky_relu(h[row] . w2 + b2, 0.01).  One wave per row.
__global__ void gate_dot(const float* __restrict__ h, const float* __restrict__ w2,
                         const float* __restrict__ b2, float* __restrict__ s, int N) {
    const int row  = blockIdx.x * 8 + (threadIdx.x >> 5);
    const int lane = threadIdx.x & 31;
    if (row >= N) return;
    const float4 hv = *(const float4*)(h  + (size_t)row * 128 + lane * 4);
    const float4 wv = *(const float4*)(w2 + lane * 4);
    float p = hv.x * wv.x + hv.y * wv.y + hv.z * wv.z + hv.w * wv.w;
    for (int o = 16; o > 0; o >>= 1) p += __shfl_xor(p, o, 32);
    if (lane == 0) {
        const float v = p + b2[0];
        s[row] = v > 0.f ? v : 0.01f * v;
    }
}

// user_out = softmax([s_inf, s_int]) . [p, q] + user_emb
__global__ void combine_user(const float* __restrict__ u, const float* __restrict__ p,
                             const float* __restrict__ q,
                             const float* __restrict__ s_inf, const float* __restrict__ s_int,
                             float* __restrict__ out, int N) {
    const int idx = blockIdx.x * blockDim.x + threadIdx.x;
    if (idx >= N * 128) return;
    const int i = idx >> 7;
    const float a = s_inf[i], b = s_int[i];
    const float m  = fmaxf(a, b);
    const float ea = __expf(a - m), eb = __expf(b - m);
    const float inv = 1.0f / (ea + eb);
    out[idx] = (ea * inv) * p[idx] + (eb * inv) * q[idx] + u[idx];
}

__global__ void fill_f32(float* __restrict__ p, float v, long long n) {
    long long i = (long long)blockIdx.x * blockDim.x + threadIdx.x;
    if (i < n) p[i] = v;
}
__global__ void fill_u32(unsigned* __restrict__ p, unsigned v, long long n) {
    long long i = (long long)blockIdx.x * blockDim.x + threadIdx.x;
    if (i < n) p[i] = v;
}
__global__ void copy_f32(float* __restrict__ d, const float* __restrict__ s, long long n) {
    long long i = (long long)blockIdx.x * blockDim.x + threadIdx.x;
    if (i < n) d[i] = s[i];
}

static inline unsigned blocks1d(long long n, int bs) { return (unsigned)((n + bs - 1) / bs); }

extern "C" void kernel_launch(void* const* d_in, const int* in_sizes, int n_in,
                              void* d_out, int out_size, void* d_ws, size_t ws_size,
                              hipStream_t stream) {
    (void)n_in; (void)out_size; (void)ws_size;
    const float* user_emb = (const float*)d_in[0];
    const float* item_emb = (const float*)d_in[1];
    const int* rate_src = (const int*)d_in[2];
    const int* rate_dst = (const int*)d_in[3];
    const int* rb_src   = (const int*)d_in[4];
    const int* rb_dst   = (const int*)d_in[5];
    const int* tr_src   = (const int*)d_in[6];
    const int* tr_dst   = (const int*)d_in[7];
    const float* rate_Wsrc = (const float*)d_in[8];
    const float* rate_bsrc = (const float*)d_in[9];
    const float* rate_Wdst = (const float*)d_in[10];
    const float* rate_bdst = (const float*)d_in[11];
    const float* rate_attn = (const float*)d_in[12];
    const float* rb_Wsrc = (const float*)d_in[13];
    const float* rb_bsrc = (const float*)d_in[14];
    const float* rb_Wdst = (const float*)d_in[15];
    const float* rb_bdst = (const float*)d_in[16];
    const float* rb_attn = (const float*)d_in[17];
    const float* tr_Wsrc = (const float*)d_in[18];
    const float* tr_bsrc = (const float*)d_in[19];
    const float* tr_Wdst = (const float*)d_in[20];
    const float* tr_bdst = (const float*)d_in[21];
    const float* tr_attn = (const float*)d_in[22];
    const float* inf_W1  = (const float*)d_in[23];
    const float* inf_b1  = (const float*)d_in[24];
    const float* inf_W2  = (const float*)d_in[25];
    const float* inf_b2  = (const float*)d_in[26];
    const float* intr_W1 = (const float*)d_in[27];
    const float* intr_b1 = (const float*)d_in[28];
    const float* intr_W2 = (const float*)d_in[29];
    const float* intr_b2 = (const float*)d_in[30];

    const int D  = 128;
    const int NU = in_sizes[0] / D;
    const int NI = in_sizes[1] / D;
    const int ER = in_sizes[2];
    const int ET = in_sizes[6];
    const int NMAX = NU > NI ? NU : NI;
    const int EMAX = ER > ET ? ER : ET;
    const size_t WPK = 16384;                 // packed bf16 elements per 128x128 W

    // ---- workspace carve (256-byte aligned regions) -------------------------
    char* base = (char*)d_ws;
    size_t off = 0;
    auto carve = [&](size_t bytes) -> char* {
        char* p = base + off;
        off += (bytes + 255) & ~(size_t)255;
        return p;
    };
    float*          fsrc   = (float*)carve((size_t)NMAX * D * 4);
    float*          fdst   = (float*)carve((size_t)NMAX * D * 4);
    float*          pbuf   = (float*)carve((size_t)NU * D * 4);
    float*          qbuf   = (float*)carve((size_t)NU * D * 4);
    float*          ebuf   = (float*)carve((size_t)EMAX * 4);
    float*          exbuf  = (float*)carve((size_t)EMAX * 4);
    unsigned*       emax   = (unsigned*)carve((size_t)NMAX * 4);
    float*          denom  = (float*)carve((size_t)NMAX * 4);
    float*          s_inf  = (float*)carve((size_t)NU * 4);
    float*          s_int  = (float*)carve((size_t)NU * 4);
    unsigned short* ubf    = (unsigned short*)carve((size_t)NU * D * 2);   // user_emb bf16
    unsigned short* ibf    = (unsigned short*)carve((size_t)NI * D * 2);   // item_emb bf16
    unsigned short* pqbf   = (unsigned short*)carve((size_t)NU * D * 2);   // p/q bf16 (reused)
    unsigned short* wpk    = (unsigned short*)carve(10 * WPK * 2);         // 10 packed W blocks
    float*          hbuf   = fsrc;  // f32 gate hidden, reused after edge phases

    unsigned short* w_rate_s = wpk + 0 * WPK;
    unsigned short* w_rate_d = wpk + 1 * WPK;
    unsigned short* w_rb_s   = wpk + 2 * WPK;
    unsigned short* w_rb_d   = wpk + 3 * WPK;
    unsigned short* w_tr_s   = wpk + 4 * WPK;
    unsigned short* w_tr_d   = wpk + 5 * WPK;
    unsigned short* w_inf_a  = wpk + 6 * WPK;
    unsigned short* w_inf_b  = wpk + 7 * WPK;
    unsigned short* w_intr_a = wpk + 8 * WPK;
    unsigned short* w_intr_b = wpk + 9 * WPK;

    float* out_user = (float*)d_out;
    float* out_item = out_user + (size_t)NU * D;

    const dim3 gU(NU / 16, 8);  // NU, NI are multiples of 16 (50000 = 3125*16)
    const dim3 gI(NI / 16, 8);

    // ---- one-shot operand preparation ---------------------------------------
    cvt_f32_bf16<<<blocks1d((long long)NU * D, 256), 256, 0, stream>>>(user_emb, ubf, (long long)NU * D);
    cvt_f32_bf16<<<blocks1d((long long)NI * D, 256), 256, 0, stream>>>(item_emb, ibf, (long long)NI * D);
    pack_w_bf16<<<4, 256, 0, stream>>>(rate_Wsrc, w_rate_s);
    pack_w_bf16<<<4, 256, 0, stream>>>(rate_Wdst, w_rate_d);
    pack_w_bf16<<<4, 256, 0, stream>>>(rb_Wsrc, w_rb_s);
    pack_w_bf16<<<4, 256, 0, stream>>>(rb_Wdst, w_rb_d);
    pack_w_bf16<<<4, 256, 0, stream>>>(tr_Wsrc, w_tr_s);
    pack_w_bf16<<<4, 256, 0, stream>>>(tr_Wdst, w_tr_d);
    pack_w_bf16<<<4, 256, 0, stream>>>(inf_W1, w_inf_a);
    pack_w_bf16<<<4, 256, 0, stream>>>(inf_W1 + 128 * 128, w_inf_b);
    pack_w_bf16<<<4, 256, 0, stream>>>(intr_W1, w_intr_a);
    pack_w_bf16<<<4, 256, 0, stream>>>(intr_W1 + 128 * 128, w_intr_b);

    // ---- relation 'rate' (user -> item), residual into out_item -------------
    gemm_wmma_bf16<<<gU, 32, 0, stream>>>(ubf, w_rate_s, nullptr, nullptr, rate_bsrc, fsrc);
    gemm_wmma_bf16<<<gI, 32, 0, stream>>>(ibf, w_rate_d, nullptr, nullptr, rate_bdst, fdst);
    fill_u32<<<blocks1d(NI, 256), 256, 0, stream>>>(emax, 0u, NI);
    fill_f32<<<blocks1d(NI, 256), 256, 0, stream>>>(denom, 0.f, NI);
    copy_f32<<<blocks1d((long long)NI * D, 256), 256, 0, stream>>>(out_item, item_emb, (long long)NI * D);
    edge_score<<<blocks1d(ER, 8), 256, 0, stream>>>(fsrc, fdst, rate_src, rate_dst, rate_attn, ebuf, emax, ER);
    edge_exp<<<blocks1d(ER, 256), 256, 0, stream>>>(ebuf, rate_dst, emax, exbuf, denom, ER);
    edge_agg<<<blocks1d(ER, 8), 256, 0, stream>>>(exbuf, denom, fsrc, rate_src, rate_dst, out_item, ER);

    // ---- relation 'rated-by' (item -> user) -> q ----------------------------
    gemm_wmma_bf16<<<gI, 32, 0, stream>>>(ibf, w_rb_s, nullptr, nullptr, rb_bsrc, fsrc);
    gemm_wmma_bf16<<<gU, 32, 0, stream>>>(ubf, w_rb_d, nullptr, nullptr, rb_bdst, fdst);
    fill_u32<<<blocks1d(NU, 256), 256, 0, stream>>>(emax, 0u, NU);
    fill_f32<<<blocks1d(NU, 256), 256, 0, stream>>>(denom, 0.f, NU);
    fill_f32<<<blocks1d((long long)NU * D, 256), 256, 0, stream>>>(qbuf, 0.f, (long long)NU * D);
    edge_score<<<blocks1d(ER, 8), 256, 0, stream>>>(fsrc, fdst, rb_src, rb_dst, rb_attn, ebuf, emax, ER);
    edge_exp<<<blocks1d(ER, 256), 256, 0, stream>>>(ebuf, rb_dst, emax, exbuf, denom, ER);
    edge_agg<<<blocks1d(ER, 8), 256, 0, stream>>>(exbuf, denom, fsrc, rb_src, rb_dst, qbuf, ER);

    // ---- relation 'trust' (user -> user) -> p -------------------------------
    gemm_wmma_bf16<<<gU, 32, 0, stream>>>(ubf, w_tr_s, nullptr, nullptr, tr_bsrc, fsrc);
    gemm_wmma_bf16<<<gU, 32, 0, stream>>>(ubf, w_tr_d, nullptr, nullptr, tr_bdst, fdst);
    fill_u32<<<blocks1d(NU, 256), 256, 0, stream>>>(emax, 0u, NU);
    fill_f32<<<blocks1d(NU, 256), 256, 0, stream>>>(denom, 0.f, NU);
    fill_f32<<<blocks1d((long long)NU * D, 256), 256, 0, stream>>>(pbuf, 0.f, (long long)NU * D);
    edge_score<<<blocks1d(ET, 8), 256, 0, stream>>>(fsrc, fdst, tr_src, tr_dst, tr_attn, ebuf, emax, ET);
    edge_exp<<<blocks1d(ET, 256), 256, 0, stream>>>(ebuf, tr_dst, emax, exbuf, denom, ET);
    edge_agg<<<blocks1d(ET, 8), 256, 0, stream>>>(exbuf, denom, fsrc, tr_src, tr_dst, pbuf, ET);

    // ---- gates: concat([u, x]) @ W1 + b1 as two K=128 WMMA passes -----------
    cvt_f32_bf16<<<blocks1d((long long)NU * D, 256), 256, 0, stream>>>(pbuf, pqbf, (long long)NU * D);
    gemm_wmma_bf16<<<gU, 32, 0, stream>>>(ubf, w_inf_a, pqbf, w_inf_b, inf_b1, hbuf);
    gate_dot<<<blocks1d(NU, 8), 256, 0, stream>>>(hbuf, inf_W2, inf_b2, s_inf, NU);
    cvt_f32_bf16<<<blocks1d((long long)NU * D, 256), 256, 0, stream>>>(qbuf, pqbf, (long long)NU * D);
    gemm_wmma_bf16<<<gU, 32, 0, stream>>>(ubf, w_intr_a, pqbf, w_intr_b, intr_b1, hbuf);
    gate_dot<<<blocks1d(NU, 8), 256, 0, stream>>>(hbuf, intr_W2, intr_b2, s_int, NU);

    // ---- gamma softmax + residual ------------------------------------------
    combine_user<<<blocks1d((long long)NU * D, 256), 256, 0, stream>>>(
        user_emb, pbuf, qbuf, s_inf, s_int, out_user, NU);
}
